// WeakSeg_2405181686459
// MI455X (gfx1250) — matrix operand order repaired and checked by reading, not compiled, not run on hardware
//
#include <hip/hip_runtime.h>
#include <math.h>

typedef __attribute__((ext_vector_type(16))) _Float16 v16h;
typedef __attribute__((ext_vector_type(8)))  float    v8f;

// ---------------------------------------------------------------------------
// WMMA GEMM: out[M,N] = A[M,K] @ W[K,N] (+bias). fp32 in/out, f16 math.
// Block = 128 threads (4 waves). Block tile = 64 rows x 64 cols:
//   wave w -> row sub-tile w (16 rows); each wave computes 4 N-sub-tiles,
//   reusing one A fragment for 4 v_wmma_f32_16x16x32_f16 per K-chunk.
// LDS holds fragment-major data so each lane loads its fragment with
// contiguous 32B (2 x ds_load_b128) reads.
// ---------------------------------------------------------------------------
__global__ __launch_bounds__(128)
void gemm_wmma_kernel(const float* __restrict__ A, const float* __restrict__ W,
                      const float* __restrict__ bias, float* __restrict__ out,
                      int M, int K, int N) {
    __shared__ __align__(32) _Float16 Af[4][32][16];   // [wave][lane][elem]
    __shared__ __align__(32) _Float16 Bf[4][32][16];   // [ntile][lane][elem]
    const int tid  = threadIdx.x;
    const int lane = tid & 31;
    const int wave = tid >> 5;
    const int rowBase = blockIdx.x * 64;
    const int colBase = blockIdx.y * 64;

    // A staging geometry for this thread (thread t stages fragment of
    // wave=t>>5, lane=t&31)
    const int mA   = lane & 15;
    const int kbA  = (lane >> 4) * 16;   // lanes 0-15: K base 0 ; 16-31: 16
    int rowA = rowBase + wave * 16 + mA;
    if (rowA > M - 1) rowA = M - 1;      // clamp: OOB rows never stored
    const float* arow = A + (size_t)rowA * K;

    // B staging geometry: thread covers lane l2, elems i0..i0+3, all 4 ntiles
    const int l2  = tid & 31;
    const int i0  = (tid >> 5) * 4;
    const int kbB = (l2 >> 4) * 16;      // lanes 0-15: K 0..15 ; 16-31: 16..31

    union { v8f v; float e[8]; } acc[4];
#pragma unroll
    for (int nt = 0; nt < 4; ++nt)
#pragma unroll
        for (int i = 0; i < 8; ++i) acc[nt].e[i] = 0.0f;

    union { v16h v; _Float16 e[16]; } af, bf;

    int kc = 0;
    // ---------------- fast path: full 32-wide K chunks ----------------
    for (; kc + 32 <= K; kc += 32) {
        const float* ap = arow + kc + kbA;
#pragma unroll
        for (int j = 0; j < 8; ++j) Af[wave][lane][j]     = (_Float16)ap[j];
#pragma unroll
        for (int j = 0; j < 8; ++j) Af[wave][lane][8 + j] = (_Float16)ap[8 + j];
#pragma unroll
        for (int nt = 0; nt < 4; ++nt) {
            int col = colBase + nt * 16 + (l2 & 15);
#pragma unroll
            for (int j = 0; j < 4; ++j) {
                int gk = kc + kbB + i0 + j;
                float v = (col < N) ? W[(size_t)gk * N + col] : 0.0f;
                Bf[nt][l2][i0 + j] = (_Float16)v;
            }
        }
        __syncthreads();
        af.v = *reinterpret_cast<const v16h*>(&Af[wave][lane][0]);
#pragma unroll
        for (int nt = 0; nt < 4; ++nt) {
            bf.v = *reinterpret_cast<const v16h*>(&Bf[nt][lane][0]);
            acc[nt].v = __builtin_amdgcn_wmma_f32_16x16x32_f16(
                false, af.v, false, bf.v, (short)0, acc[nt].v, false, false);
        }
        __syncthreads();
    }
    // ---------------- tail chunk (K not multiple of 32) ----------------
    if (kc < K) {
        const int krem = K - kc;
#pragma unroll
        for (int j = 0; j < 8; ++j) {
            int off = kbA + j;
            Af[wave][lane][j] = (_Float16)((off < krem) ? arow[kc + off] : 0.0f);
        }
#pragma unroll
        for (int j = 0; j < 8; ++j) {
            int off = kbA + 8 + j;
            Af[wave][lane][8 + j] = (_Float16)((off < krem) ? arow[kc + off] : 0.0f);
        }
#pragma unroll
        for (int nt = 0; nt < 4; ++nt) {
            int col = colBase + nt * 16 + (l2 & 15);
#pragma unroll
            for (int j = 0; j < 4; ++j) {
                int off = kbB + i0 + j;
                float v = (off < krem && col < N) ? W[(size_t)(kc + off) * N + col] : 0.0f;
                Bf[nt][l2][i0 + j] = (_Float16)v;
            }
        }
        __syncthreads();
        af.v = *reinterpret_cast<const v16h*>(&Af[wave][lane][0]);
#pragma unroll
        for (int nt = 0; nt < 4; ++nt) {
            bf.v = *reinterpret_cast<const v16h*>(&Bf[nt][lane][0]);
            acc[nt].v = __builtin_amdgcn_wmma_f32_16x16x32_f16(
                false, af.v, false, bf.v, (short)0, acc[nt].v, false, false);
        }
        __syncthreads();
    }

    // C/D layout: VGPR r -> M = r + 8*(lane>=16), N = lane&15
#pragma unroll
    for (int nt = 0; nt < 4; ++nt) {
#pragma unroll
        for (int r = 0; r < 8; ++r) {
            int gm = rowBase + wave * 16 + (lane >> 4) * 8 + r;
            int gn = colBase + nt * 16 + (lane & 15);
            if (gm < M && gn < N) {
                float v = acc[nt].e[r];
                if (bias) v += bias[gn];
                out[(size_t)gm * N + gn] = v;
            }
        }
    }
}

// ---------------------------------------------------------------------------
// BatchNorm (training-mode batch stats) over rows of X[R,C]
// ---------------------------------------------------------------------------
__global__ __launch_bounds__(256)
void bn_stats_kernel(const float* __restrict__ X, int R, int C,
                     float* __restrict__ mean, float* __restrict__ istd) {
    __shared__ float s1[256], s2[256];
    int c = blockIdx.x;
    int tid = threadIdx.x;
    float a = 0.0f, b = 0.0f;
    for (int r = tid; r < R; r += 256) {
        float v = X[r * C + c];
        a += v; b += v * v;
    }
    s1[tid] = a; s2[tid] = b;
    __syncthreads();
    for (int o = 128; o > 0; o >>= 1) {
        if (tid < o) { s1[tid] += s1[tid + o]; s2[tid] += s2[tid + o]; }
        __syncthreads();
    }
    if (tid == 0) {
        float m = s1[0] / (float)R;
        float v = s2[0] / (float)R - m * m;
        mean[c] = m;
        istd[c] = rsqrtf(v + 1e-5f);
    }
}

__global__ void bn_apply_kernel(float* __restrict__ X, int R, int C,
                                const float* __restrict__ mean,
                                const float* __restrict__ istd,
                                const float* __restrict__ g,
                                const float* __restrict__ b, int relu) {
    int i = blockIdx.x * blockDim.x + threadIdx.x;
    if (i >= R * C) return;
    int c = i % C;
    float v = (X[i] - mean[c]) * istd[c] * g[c] + b[c];
    if (relu) v = fmaxf(v, 0.0f);
    X[i] = v;
}

// ---------------------------------------------------------------------------
// FPS: single block, sequential argmax scan (matches pointops semantics)
// ---------------------------------------------------------------------------
__global__ __launch_bounds__(1024)
void fps_kernel(const float* __restrict__ p, int Np, int m,
                int* __restrict__ idx, float* __restrict__ dists) {
    __shared__ float bd[1024];
    __shared__ int   bi[1024];
    __shared__ int   s_last;
    int tid = threadIdx.x;
    for (int i = tid; i < Np; i += 1024) dists[i] = 3.4e38f;
    if (tid == 0) { idx[0] = 0; s_last = 0; }
    __syncthreads();
    for (int it = 1; it < m; ++it) {
        int last = s_last;
        float lx = p[3 * last], ly = p[3 * last + 1], lz = p[3 * last + 2];
        float best = -1.0f; int besti = 0;
        for (int i = tid; i < Np; i += 1024) {
            float dx = p[3 * i] - lx, dy = p[3 * i + 1] - ly, dz = p[3 * i + 2] - lz;
            float d = dx * dx + dy * dy + dz * dz;
            float dd = fminf(dists[i], d);
            dists[i] = dd;
            if (dd > best) { best = dd; besti = i; }
        }
        bd[tid] = best; bi[tid] = besti;
        __syncthreads();
        for (int o = 512; o > 0; o >>= 1) {
            if (tid < o && bd[tid + o] > bd[tid]) { bd[tid] = bd[tid + o]; bi[tid] = bi[tid + o]; }
            __syncthreads();
        }
        if (tid == 0) { idx[it] = bi[0]; s_last = bi[0]; }
        __syncthreads();
    }
}

// ---------------------------------------------------------------------------
// kNN (k=16): one thread per query, insertion sort in registers
// ---------------------------------------------------------------------------
template <int KN>
__global__ void knn_kernel(const float* __restrict__ q, int Mq,
                           const float* __restrict__ s, int Ms,
                           int* __restrict__ out) {
    int i = blockIdx.x * blockDim.x + threadIdx.x;
    if (i >= Mq) return;
    float qx = q[3 * i], qy = q[3 * i + 1], qz = q[3 * i + 2];
    float bd[KN]; int bi[KN];
#pragma unroll
    for (int j = 0; j < KN; ++j) { bd[j] = 3.4e38f; bi[j] = 0; }
    for (int j = 0; j < Ms; ++j) {
        float dx = s[3 * j] - qx, dy = s[3 * j + 1] - qy, dz = s[3 * j + 2] - qz;
        float d = dx * dx + dy * dy + dz * dz;
        if (d < bd[KN - 1]) {
            int pos = KN - 1;
            while (pos > 0 && bd[pos - 1] > d) {
                bd[pos] = bd[pos - 1]; bi[pos] = bi[pos - 1]; --pos;
            }
            bd[pos] = d; bi[pos] = j;
        }
    }
    for (int j = 0; j < KN; ++j) out[i * KN + j] = bi[j];
}

// grouped[m*ns+s][c] = c<3 ? p_prev[nidx]-np : x_prev[nidx]
__global__ void group_kernel(const float* __restrict__ p_prev,
                             const float* __restrict__ x_prev,
                             const float* __restrict__ np_,
                             const int* __restrict__ nidx,
                             int Mq, int ns, int C, float* __restrict__ grouped) {
    int Ctot = 3 + C;
    int total = Mq * ns * Ctot;
    int i = blockIdx.x * blockDim.x + threadIdx.x;
    if (i >= total) return;
    int c = i % Ctot;
    int row = i / Ctot;
    int m = row / ns;
    int j = nidx[row];
    float v;
    if (c < 3) v = p_prev[3 * j + c] - np_[3 * m + c];
    else       v = x_prev[j * C + (c - 3)];
    grouped[i] = v;
}

__global__ void gather3_kernel(const float* __restrict__ p, const int* __restrict__ idx,
                               float* __restrict__ out, int m) {
    int i = blockIdx.x * blockDim.x + threadIdx.x;
    if (i >= m) return;
    int j = idx[i];
    out[3 * i] = p[3 * j]; out[3 * i + 1] = p[3 * j + 1]; out[3 * i + 2] = p[3 * j + 2];
}

__global__ void maxpool_kernel(const float* __restrict__ h, int M, int ns, int C,
                               float* __restrict__ out) {
    int i = blockIdx.x * blockDim.x + threadIdx.x;
    if (i >= M * C) return;
    int m = i / C, c = i % C;
    float best = -3.4e38f;
    for (int s = 0; s < ns; ++s) best = fmaxf(best, h[(m * ns + s) * C + c]);
    out[i] = best;
}

__global__ void concat6_kernel(const float* __restrict__ x0, const float* __restrict__ p0,
                               float* __restrict__ xin, int Nn) {
    int i = blockIdx.x * blockDim.x + threadIdx.x;
    if (i >= Nn) return;
    xin[i * 6 + 0] = x0[3 * i];     xin[i * 6 + 1] = x0[3 * i + 1]; xin[i * 6 + 2] = x0[3 * i + 2];
    xin[i * 6 + 3] = p0[3 * i];     xin[i * 6 + 4] = p0[3 * i + 1]; xin[i * 6 + 5] = p0[3 * i + 2];
}

__global__ void colmean_kernel(const float* __restrict__ X, int R, int C,
                               float* __restrict__ out) {
    int c = blockIdx.x * blockDim.x + threadIdx.x;
    if (c >= C) return;
    float s = 0.0f;
    for (int r = 0; r < R; ++r) s += X[r * C + c];
    out[c] = s / (float)R;
}

__global__ void vecfc_relu_kernel(const float* __restrict__ x, const float* __restrict__ W,
                                  const float* __restrict__ b, float* __restrict__ out,
                                  int K, int N) {
    int j = blockIdx.x * blockDim.x + threadIdx.x;
    if (j >= N) return;
    float s = b[j];
    for (int k = 0; k < K; ++k) s += x[k] * W[k * N + j];
    out[j] = fmaxf(s, 0.0f);
}

__global__ void concat_bcast_kernel(const float* __restrict__ xa, const float* __restrict__ gf,
                                    float* __restrict__ out, int R, int C1, int C2) {
    int Ct = C1 + C2;
    int i = blockIdx.x * blockDim.x + threadIdx.x;
    if (i >= R * Ct) return;
    int r = i / Ct, c = i % Ct;
    out[i] = (c < C1) ? xa[r * C1 + c] : gf[c - C1];
}

// out[i,:] += 3-NN inverse-distance weighted sum of x_src
__global__ void interp_add_kernel(const float* __restrict__ p_src,
                                  const float* __restrict__ x_src, int Ms,
                                  const float* __restrict__ p_dst, int Md, int C,
                                  float* __restrict__ out) {
    int i = blockIdx.x * blockDim.x + threadIdx.x;
    if (i >= Md) return;
    float qx = p_dst[3 * i], qy = p_dst[3 * i + 1], qz = p_dst[3 * i + 2];
    float bd0 = 3.4e38f, bd1 = 3.4e38f, bd2 = 3.4e38f;
    int b0 = 0, b1 = 0, b2 = 0;
    for (int j = 0; j < Ms; ++j) {
        float dx = p_src[3 * j] - qx, dy = p_src[3 * j + 1] - qy, dz = p_src[3 * j + 2] - qz;
        float d = dx * dx + dy * dy + dz * dz;
        if (d < bd2) {
            if (d < bd0)      { bd2 = bd1; b2 = b1; bd1 = bd0; b1 = b0; bd0 = d; b0 = j; }
            else if (d < bd1) { bd2 = bd1; b2 = b1; bd1 = d; b1 = j; }
            else              { bd2 = d; b2 = j; }
        }
    }
    float w0 = 1.0f / (sqrtf(bd0) + 1e-8f);
    float w1 = 1.0f / (sqrtf(bd1) + 1e-8f);
    float w2 = 1.0f / (sqrtf(bd2) + 1e-8f);
    float ws = w0 + w1 + w2;
    w0 /= ws; w1 /= ws; w2 /= ws;
    for (int c = 0; c < C; ++c)
        out[i * C + c] += w0 * x_src[b0 * C + c] + w1 * x_src[b1 * C + c] + w2 * x_src[b2 * C + c];
}

// out[M,N] = h[M,K] @ W[K,N] + b  (final classifier, N=13)
__global__ void final_kernel(const float* __restrict__ h, const float* __restrict__ W,
                             const float* __restrict__ b, float* __restrict__ out,
                             int M, int K, int N) {
    int i = blockIdx.x * blockDim.x + threadIdx.x;
    if (i >= M * N) return;
    int r = i / N, c = i % N;
    float s = b[c];
    for (int k = 0; k < K; ++k) s += h[r * K + k] * W[k * N + c];
    out[i] = s;
}

// ---------------------------------------------------------------------------
// Host orchestration
// ---------------------------------------------------------------------------
static inline void run_gemm(const float* A, const float* W, const float* bias,
                            float* out, int M, int K, int N, hipStream_t s) {
    dim3 g((M + 63) / 64, (N + 63) / 64);
    gemm_wmma_kernel<<<g, 128, 0, s>>>(A, W, bias, out, M, K, N);
}

static inline void run_bn(float* X, int R, int C, const float* g, const float* b,
                          float* mean, float* istd, int relu, hipStream_t s) {
    bn_stats_kernel<<<C, 256, 0, s>>>(X, R, C, mean, istd);
    int total = R * C;
    bn_apply_kernel<<<(total + 255) / 256, 256, 0, s>>>(X, R, C, mean, istd, g, b, relu);
}

extern "C" void kernel_launch(void* const* d_in, const int* in_sizes, int n_in,
                              void* d_out, int out_size, void* d_ws, size_t ws_size,
                              hipStream_t stream) {
    const int N = 16384;
    const int NS = 16;
    const float* p0 = (const float*)d_in[0];
    const float* x0 = (const float*)d_in[1];
    // params: sorted weight names, then BN params in dict-insertion order
    const float* clsW1  = (const float*)d_in[2];
    const float* clsW2  = (const float*)d_in[3];
    const float* d1_l1W = (const float*)d_in[4];
    const float* d1_l2W = (const float*)d_in[5];
    const float* d2_l1W = (const float*)d_in[6];
    const float* d2_l2W = (const float*)d_in[7];
    const float* d3_l1W = (const float*)d_in[8];
    const float* d3_l2W = (const float*)d_in[9];
    const float* d4_l1W = (const float*)d_in[10];
    const float* d4_l2W = (const float*)d_in[11];
    const float* d5_l1W = (const float*)d_in[12];
    const float* d5_l2W = (const float*)d_in[13];
    const float* enc_W1 = (const float*)d_in[14];
    const float* enc_W2 = (const float*)d_in[15];
    const float* enc_W3 = (const float*)d_in[16];
    const float* enc_W4 = (const float*)d_in[17];
    const float* enc_W5 = (const float*)d_in[18];
    const float* enc_g1  = (const float*)d_in[19];
    const float* enc_be1 = (const float*)d_in[20];
    const float* enc_g2  = (const float*)d_in[21];
    const float* enc_be2 = (const float*)d_in[22];
    const float* enc_g3  = (const float*)d_in[23];
    const float* enc_be3 = (const float*)d_in[24];
    const float* enc_g4  = (const float*)d_in[25];
    const float* enc_be4 = (const float*)d_in[26];
    const float* enc_g5  = (const float*)d_in[27];
    const float* enc_be5 = (const float*)d_in[28];
    const float* d5_l2b  = (const float*)d_in[29];
    const float* d5_l1b  = (const float*)d_in[30];
    const float* d5_g    = (const float*)d_in[31];
    const float* d5_be   = (const float*)d_in[32];
    const float* d4_l1b  = (const float*)d_in[33];
    const float* d4_l2b  = (const float*)d_in[34];
    const float* d4_g1   = (const float*)d_in[35];
    const float* d4_be1  = (const float*)d_in[36];
    const float* d4_g2   = (const float*)d_in[37];
    const float* d4_be2  = (const float*)d_in[38];
    const float* d3_l1b  = (const float*)d_in[39];
    const float* d3_l2b  = (const float*)d_in[40];
    const float* d3_g1   = (const float*)d_in[41];
    const float* d3_be1  = (const float*)d_in[42];
    const float* d3_g2   = (const float*)d_in[43];
    const float* d3_be2  = (const float*)d_in[44];
    const float* d2_l1b  = (const float*)d_in[45];
    const float* d2_l2b  = (const float*)d_in[46];
    const float* d2_g1   = (const float*)d_in[47];
    const float* d2_be1  = (const float*)d_in[48];
    const float* d2_g2   = (const float*)d_in[49];
    const float* d2_be2  = (const float*)d_in[50];
    const float* d1_l1b  = (const float*)d_in[51];
    const float* d1_l2b  = (const float*)d_in[52];
    const float* d1_g1   = (const float*)d_in[53];
    const float* d1_be1  = (const float*)d_in[54];
    const float* d1_g2   = (const float*)d_in[55];
    const float* d1_be2  = (const float*)d_in[56];
    const float* clsb1   = (const float*)d_in[57];
    const float* cls_g   = (const float*)d_in[58];
    const float* cls_be  = (const float*)d_in[59];
    const float* clsb2   = (const float*)d_in[60];

    // --- workspace bump allocator ---
    char* ws = (char*)d_ws;
    size_t off = 0;
    auto allocf = [&](size_t n) -> float* {
        float* p = (float*)(ws + off);
        off += ((n * sizeof(float) + 255) / 256) * 256;
        return p;
    };
    auto alloci = [&](size_t n) -> int* {
        int* p = (int*)(ws + off);
        off += ((n * sizeof(int) + 255) / 256) * 256;
        return p;
    };

    float* xin   = allocf((size_t)N * 6);
    float* x1    = allocf((size_t)N * 32);
    float* G     = allocf((size_t)4096 * 16 * 35);  // grouped (reused, max size)
    float* H     = allocf((size_t)4096 * 16 * 64);  // pre-pool features (reused, max)
    float* p2    = allocf(4096 * 3);
    float* p3    = allocf(1024 * 3);
    float* p4    = allocf(256 * 3);
    float* p5    = allocf(64 * 3);
    float* x2    = allocf(4096 * 64);
    float* x3    = allocf(1024 * 128);
    float* x4    = allocf(256 * 256);
    float* x5    = allocf(64 * 512);
    int*   idx_s = alloci(4096);
    int*   nidx  = alloci(4096 * 16);
    float* dists = allocf(16384);
    float* meanb = allocf(512);
    float* istdb = allocf(512);
    float* gmean = allocf(512);
    float* gfeat = allocf(512);
    float* x5cat = allocf(64 * 1024);
    float* x5n   = allocf(64 * 512);
    float* a4    = allocf(256 * 256);
    float* c4    = allocf(64 * 256);
    float* a3    = allocf(1024 * 128);
    float* c3    = allocf(256 * 128);
    float* a2    = allocf(4096 * 64);
    float* c2    = allocf(1024 * 64);
    float* a1    = allocf((size_t)N * 32);
    float* c1    = allocf(4096 * 32);
    float* chls  = allocf((size_t)N * 32);
    (void)ws_size; (void)n_in; (void)in_sizes; (void)out_size;

    // ---------------- encoder level 1 ----------------
    concat6_kernel<<<(N + 255) / 256, 256, 0, stream>>>(x0, p0, xin, N);
    run_gemm(xin, enc_W1, nullptr, x1, N, 6, 32, stream);
    run_bn(x1, N, 32, enc_g1, enc_be1, meanb, istdb, 1, stream);

    // level descriptors
    const float* p_prev_arr[4] = { p0, p2, p3, p4 };
    float*       x_prev_arr[4] = { x1, x2, x3, x4 };
    const int    Mprev_arr[4]  = { 16384, 4096, 1024, 256 };
    const int    Cprev_arr[4]  = { 32, 64, 128, 256 };
    const int    M_arr[4]      = { 4096, 1024, 256, 64 };
    const int    Cout_arr[4]   = { 64, 128, 256, 512 };
    float*       p_s_arr[4]    = { p2, p3, p4, p5 };
    float*       x_out_arr[4]  = { x2, x3, x4, x5 };
    const float* W_arr[4]      = { enc_W2, enc_W3, enc_W4, enc_W5 };
    const float* g_arr[4]      = { enc_g2, enc_g3, enc_g4, enc_g5 };
    const float* be_arr[4]     = { enc_be2, enc_be3, enc_be4, enc_be5 };

    for (int l = 0; l < 4; ++l) {
        const float* pp = p_prev_arr[l];
        const float* xp = x_prev_arr[l];
        int Mp = Mprev_arr[l], Cp = Cprev_arr[l], Mc = M_arr[l], Co = Cout_arr[l];
        int Cin = 3 + Cp;
        float* ps = p_s_arr[l];
        fps_kernel<<<1, 1024, 0, stream>>>(pp, Mp, Mc, idx_s, dists);
        gather3_kernel<<<(Mc + 255) / 256, 256, 0, stream>>>(pp, idx_s, ps, Mc);
        knn_kernel<16><<<(Mc + 63) / 64, 64, 0, stream>>>(ps, Mc, pp, Mp, nidx);
        {
            int total = Mc * NS * Cin;
            group_kernel<<<(total + 255) / 256, 256, 0, stream>>>(pp, xp, ps, nidx, Mc, NS, Cp, G);
        }
        run_gemm(G, W_arr[l], nullptr, H, Mc * NS, Cin, Co, stream);
        run_bn(H, Mc * NS, Co, g_arr[l], be_arr[l], meanb, istdb, 1, stream);
        maxpool_kernel<<<(Mc * Co + 255) / 256, 256, 0, stream>>>(H, Mc, NS, Co, x_out_arr[l]);
    }

    // ---------------- decoder head (dec5) ----------------
    colmean_kernel<<<2, 256, 0, stream>>>(x5, 64, 512, gmean);
    vecfc_relu_kernel<<<2, 256, 0, stream>>>(gmean, d5_l2W, d5_l2b, gfeat, 512, 512);
    concat_bcast_kernel<<<(64 * 1024 + 255) / 256, 256, 0, stream>>>(x5, gfeat, x5cat, 64, 512, 512);
    run_gemm(x5cat, d5_l1W, d5_l1b, x5n, 64, 1024, 512, stream);
    run_bn(x5n, 64, 512, d5_g, d5_be, meanb, istdb, 1, stream);

    // ---------------- fp4 ----------------
    run_gemm(x4, d4_l1W, d4_l1b, a4, 256, 256, 256, stream);
    run_bn(a4, 256, 256, d4_g1, d4_be1, meanb, istdb, 1, stream);
    run_gemm(x5n, d4_l2W, d4_l2b, c4, 64, 512, 256, stream);
    run_bn(c4, 64, 256, d4_g2, d4_be2, meanb, istdb, 1, stream);
    interp_add_kernel<<<(256 + 63) / 64, 64, 0, stream>>>(p5, c4, 64, p4, 256, 256, a4);

    // ---------------- fp3 ----------------
    run_gemm(x3, d3_l1W, d3_l1b, a3, 1024, 128, 128, stream);
    run_bn(a3, 1024, 128, d3_g1, d3_be1, meanb, istdb, 1, stream);
    run_gemm(a4, d3_l2W, d3_l2b, c3, 256, 256, 128, stream);
    run_bn(c3, 256, 128, d3_g2, d3_be2, meanb, istdb, 1, stream);
    interp_add_kernel<<<(1024 + 63) / 64, 64, 0, stream>>>(p4, c3, 256, p3, 1024, 128, a3);

    // ---------------- fp2 ----------------
    run_gemm(x2, d2_l1W, d2_l1b, a2, 4096, 64, 64, stream);
    run_bn(a2, 4096, 64, d2_g1, d2_be1, meanb, istdb, 1, stream);
    run_gemm(a3, d2_l2W, d2_l2b, c2, 1024, 128, 64, stream);
    run_bn(c2, 1024, 64, d2_g2, d2_be2, meanb, istdb, 1, stream);
    interp_add_kernel<<<(4096 + 63) / 64, 64, 0, stream>>>(p3, c2, 1024, p2, 4096, 64, a2);

    // ---------------- fp1 ----------------
    run_gemm(x1, d1_l1W, d1_l1b, a1, N, 32, 32, stream);
    run_bn(a1, N, 32, d1_g1, d1_be1, meanb, istdb, 1, stream);
    run_gemm(a2, d1_l2W, d1_l2b, c1, 4096, 64, 32, stream);
    run_bn(c1, 4096, 32, d1_g2, d1_be2, meanb, istdb, 1, stream);
    interp_add_kernel<<<(N + 63) / 64, 64, 0, stream>>>(p2, c1, 4096, p0, N, 32, a1);

    // ---------------- classifier ----------------
    run_gemm(a1, clsW1, clsb1, chls, N, 32, 32, stream);
    run_bn(chls, N, 32, cls_g, cls_be, meanb, istdb, 1, stream);
    final_kernel<<<(N * 13 + 255) / 256, 256, 0, stream>>>(chls, clsW2, clsb2,
                                                           (float*)d_out, N, 32, 13);
}
